// MultiHeadedAttention_74689481277507
// MI455X (gfx1250) — matrix operand order repaired and checked by reading, not compiled, not run on hardware
//
#include <hip/hip_runtime.h>
#include <hip/hip_bf16.h>

#define BB 2
#define SS 2048
#define DD 1024
#define HH 16
#define DKK 64

typedef __attribute__((ext_vector_type(16))) _Float16 v16h;
typedef __attribute__((ext_vector_type(8)))  _Float16 v8h;
typedef __attribute__((ext_vector_type(8)))  float    v8f;
typedef __attribute__((ext_vector_type(4)))  float    v4f;
typedef __attribute__((ext_vector_type(4)))  int      v4i;

// ---------------------------------------------------------------------------
// gfx1250 async global->LDS staging (ASYNCcnt-tracked). The builtin takes
// (v4i addrspace(1)* src, v4i addrspace(3)* dst, imm offset, imm cpol).
// Falls back to synchronous LDS stores when the builtin isn't exposed.
// ---------------------------------------------------------------------------
#if defined(__has_builtin)
#if __has_builtin(__builtin_amdgcn_global_load_async_to_lds_b128)
#define HAVE_ASYNC_LDS 1
#endif
#endif

#ifdef HAVE_ASYNC_LDS
typedef __attribute__((address_space(1))) v4i* gas_b128_t;
typedef __attribute__((address_space(3))) v4i* las_b128_t;
#define ASYNC_COPY_B128(dst, src)                                              \
    __builtin_amdgcn_global_load_async_to_lds_b128(                            \
        (gas_b128_t)(uintptr_t)(src),                                          \
        (las_b128_t)(uint32_t)(uintptr_t)(dst), 0, 0)
#if __has_builtin(__builtin_amdgcn_s_wait_asynccnt)
#define ASYNC_WAIT() __builtin_amdgcn_s_wait_asynccnt(0)
#else
#define ASYNC_WAIT() asm volatile("s_wait_asynccnt 0" ::: "memory")
#endif
#else
#define ASYNC_COPY_B128(dst, src) (*(v8h*)(dst) = *(const v8h*)(src))
#define ASYNC_WAIT() ((void)0)
#endif

// ---------------------------------------------------------------------------
// WMMA fragment loaders (v_wmma_f32_16x16x32_f16 operand layouts, wave32)
// A 16x32 f16: lane 0-15 = row M=lane, halves 0..7 = K kb..kb+7, halves
// 8..15 = K kb+16..kb+23 with kb = 0 (lanes<16) / 8 (lanes>=16).
// ---------------------------------------------------------------------------
__device__ inline v16h load_frag_a(const _Float16* base, int ldk, int kofs, int lane) {
    const int m  = lane & 15;
    const int kb = kofs + ((lane & 16) ? 8 : 0);
    const _Float16* p = base + m * ldk + kb;
    v8h lo = *(const v8h*)(p);
    v8h hi = *(const v8h*)(p + 16);
    v16h r;
#pragma unroll
    for (int i = 0; i < 8; ++i) { r[i] = lo[i]; r[i + 8] = hi[i]; }
    return r;
}

// B 32x16 f16 (stored here as row-major [N][K] tiles): lane 0-15 = col N=lane,
// halves 0..7 = K kb..kb+7, halves 8..15 = K kb+16..kb+23, kb = 0 (lanes<16) / 8 (lanes>=16).
__device__ inline v16h load_frag_b(const _Float16* base, int ldk, int kofs, int lane) {
    const int n  = lane & 15;
    const int kb = kofs + ((lane & 16) ? 8 : 0);
    const _Float16* p = base + n * ldk + kb;
    v8h lo = *(const v8h*)(p);
    v8h hi = *(const v8h*)(p + 16);
    v16h r;
#pragma unroll
    for (int i = 0; i < 8; ++i) { r[i] = lo[i]; r[i + 8] = hi[i]; }
    return r;
}

__device__ inline v8f wmma_f16(v16h a, v16h b, v8f c) {
    return __builtin_amdgcn_wmma_f32_16x16x32_f16(false, a, false, b, (short)0, c,
                                                  false, false);
}

// ---------------------------------------------------------------------------
// GEMM: out[4096,1024] = X[4096,1024] * W[1024,1024] + bias
// mode 0: -> f16  |  mode 1: -> RoPE -> f16  |  mode 2: -> f32
// Tile 128x64 per 256-thread WG (8 waves, each wave a 16x64 strip).
// K-step 64 => 8 WMMAs per barrier pair per wave.
// ---------------------------------------------------------------------------
__global__ __launch_bounds__(256)
void proj_gemm(const float* __restrict__ X, const float* __restrict__ W,
               const float* __restrict__ bias, const float* __restrict__ mz,
               _Float16* __restrict__ out16, float* __restrict__ out32, int mode) {
    __shared__ _Float16 As[128 * 72];   // [row][k], padded stride 72 halves
    __shared__ _Float16 Bt[64 * 72];    // [n][k] (W transposed tile)

    const int tid  = threadIdx.x;
    const int lane = tid & 31;
    const int wid  = tid >> 5;
    const int n0   = blockIdx.x * 64;
    const int m0   = blockIdx.y * 128;

    v8f acc[4];
#pragma unroll
    for (int j = 0; j < 4; ++j)
#pragma unroll
        for (int i = 0; i < 8; ++i) acc[j][i] = 0.0f;

    for (int k0 = 0; k0 < DD; k0 += 64) {
        // stage A tile 128x64 (f32 -> f16)
#pragma unroll
        for (int i = 0; i < 8; ++i) {
            int idx = tid + 256 * i;          // 2048 float4 chunks
            int row = idx >> 4;
            int c4  = idx & 15;
            v4f v = *(const v4f*)(X + (size_t)(m0 + row) * DD + k0 + c4 * 4);
            _Float16* dst = &As[row * 72 + c4 * 4];
#pragma unroll
            for (int jj = 0; jj < 4; ++jj) dst[jj] = (_Float16)v[jj];
        }
        // stage Bt tile 64x64: Bt[n][k] = W[k0+k][n0+n]
#pragma unroll
        for (int i = 0; i < 4; ++i) {
            int idx = tid + 256 * i;          // 1024 float4 chunks
            int kk  = idx >> 4;
            int c4  = idx & 15;
            v4f v = *(const v4f*)(W + (size_t)(k0 + kk) * DD + n0 + c4 * 4);
#pragma unroll
            for (int jj = 0; jj < 4; ++jj)
                Bt[(c4 * 4 + jj) * 72 + kk] = (_Float16)v[jj];
        }
        // prefetch next-stage tiles into cache (global_prefetch_b8)
        if (k0 + 64 < DD) {
            __builtin_prefetch(X + (size_t)(m0 + (tid >> 4)) * DD + k0 + 64 + (tid & 15) * 4, 0, 1);
            __builtin_prefetch(W + (size_t)(k0 + 64 + (tid >> 4)) * DD + n0 + (tid & 15) * 4, 0, 1);
        }
        __syncthreads();

#pragma unroll
        for (int kk = 0; kk < 64; kk += 32) {
            v16h a = load_frag_a(&As[(wid * 16) * 72], 72, kk, lane);
            v16h b[4];
#pragma unroll
            for (int j = 0; j < 4; ++j)
                b[j] = load_frag_b(&Bt[(j * 16) * 72], 72, kk, lane);
#pragma unroll
            for (int j = 0; j < 4; ++j)
                acc[j] = wmma_f16(a, b[j], acc[j]);
        }
        __syncthreads();
    }

    // epilogue: bias (+ fused RoPE) + store
    const int l   = lane & 15;
    const int hi8 = (lane & 16) ? 8 : 0;
#pragma unroll
    for (int j = 0; j < 4; ++j) {
#pragma unroll
        for (int r = 0; r < 8; ++r) {
            const int gm  = m0 + wid * 16 + r + hi8;     // row in [B*S)
            const int gn  = n0 + j * 16 + l;             // col in [D)
            float val = acc[j][r] + bias[gn];
            if (mode == 1) {
                // d = gn % 64 = j*16 + l  (n0 is 64-aligned)
                const int d   = j * 16 + l;
                const int gnp = (j < 2) ? gn + 32 : gn - 32;
                float pv  = acc[j ^ 2][r] + bias[gnp];
                float rot = (j < 2) ? -pv : pv;          // rotate_half
                float pos = mz[gm];                      // mz is [B,S] flat
                // theta_i = 10000^(-i/32), i = d>>1 ; ln(10000)/32
                float inv = __expf(-0.28782313662425575f * (float)(d >> 1));
                float sn, cs;
                __sincosf(pos * inv, &sn, &cs);
                val = val * cs + rot * sn;
            }
            if (mode == 2) out32[(size_t)gm * DD + gn] = val;
            else           out16[(size_t)gm * DD + gn] = (_Float16)val;
        }
    }
}

// ---------------------------------------------------------------------------
// Flash attention: one WG per (b, h, 128-query tile); 8 waves x 16 q-rows.
// ---------------------------------------------------------------------------
__global__ __launch_bounds__(256)
void attention(const _Float16* __restrict__ Q, const _Float16* __restrict__ K,
               const _Float16* __restrict__ V, const int* __restrict__ mask,
               float* __restrict__ O) {
    __shared__ _Float16 Qs[128 * 72];       // [qrow][dk]
    __shared__ _Float16 Ks[64 * 72];        // [key][dk]
    __shared__ _Float16 Vt[64 * 72];        // [dk][key]
    __shared__ _Float16 Ps[8 * 16 * 72];    // per-wave P strip [row][key]

    const int tid  = threadIdx.x;
    const int lane = tid & 31;
    const int wid  = tid >> 5;
    const int tq   = blockIdx.x & 15;           // S/128 = 16 tiles
    const int bh   = blockIdx.x >> 4;
    const int b    = bh >> 4;
    const int h    = bh & 15;
    const int q0   = b * SS + tq * 128;         // row base in [B*S)
    const size_t hoff = (size_t)h * DKK;

    // load Q tile 128x64 into LDS (async-to-LDS when available)
#pragma unroll
    for (int i = 0; i < 4; ++i) {
        int idx = tid + 256 * i;                // 1024 8-half chunks
        int row = idx >> 3, c = idx & 7;
        ASYNC_COPY_B128(&Qs[row * 72 + c * 8],
                        Q + (size_t)(q0 + row) * DD + hoff + c * 8);
    }

    float M[8], L[8];
    v8f o[4];
#pragma unroll
    for (int r = 0; r < 8; ++r) { M[r] = -3.0e38f; L[r] = 0.0f; }
#pragma unroll
    for (int j = 0; j < 4; ++j)
#pragma unroll
        for (int i = 0; i < 8; ++i) o[j][i] = 0.0f;

    const int l   = lane & 15;
    const int hi8 = (lane & 16) ? 8 : 0;
    _Float16* Pw  = &Ps[wid * 16 * 72];
    const int qbase = tq * 128 + wid * 16;      // query index within S

    for (int kt = 0; kt < SS; kt += 64) {
        __syncthreads();                        // readers of prev K/V done
        // stage K tile 64x64 (async-to-LDS when available)
#pragma unroll
        for (int i = 0; i < 2; ++i) {
            int idx = tid + 256 * i;
            int row = idx >> 3, c = idx & 7;
            ASYNC_COPY_B128(&Ks[row * 72 + c * 8],
                            K + (size_t)(b * SS + kt + row) * DD + hoff + c * 8);
        }
        // stage V tile transposed: Vt[dk][key] (manual transpose via ds stores)
#pragma unroll
        for (int i = 0; i < 2; ++i) {
            int idx = tid + 256 * i;
            int key = idx >> 3, c = idx & 7;
            v8h v = *(const v8h*)(V + (size_t)(b * SS + kt + key) * DD + hoff + c * 8);
#pragma unroll
            for (int jj = 0; jj < 8; ++jj) Vt[(c * 8 + jj) * 72 + key] = v[jj];
        }
        if (kt + 64 < SS)                        // prefetch next K tile
            __builtin_prefetch(K + (size_t)(b * SS + kt + 64 + (tid >> 2)) * DD + hoff, 0, 1);
        ASYNC_WAIT();
        __syncthreads();

        // scores S = Q * K^T  (16 q-rows x 64 keys per wave)
        v8f s[4];
#pragma unroll
        for (int j = 0; j < 4; ++j)
#pragma unroll
            for (int i = 0; i < 8; ++i) s[j][i] = 0.0f;
#pragma unroll
        for (int kk = 0; kk < 64; kk += 32) {
            v16h a = load_frag_a(&Qs[(wid * 16) * 72], 72, kk, lane);
#pragma unroll
            for (int j = 0; j < 4; ++j) {
                v16h bf = load_frag_b(&Ks[(j * 16) * 72], 72, kk, lane);
                s[j] = wmma_f16(a, bf, s[j]);
            }
        }
        // mask (reference: mask==0 -> -1e9)
#pragma unroll
        for (int j = 0; j < 4; ++j) {
            const int keyg = kt + j * 16 + l;
#pragma unroll
            for (int r = 0; r < 8; ++r) {
                const int qg = qbase + r + hi8;
                if (mask[((size_t)b * SS + qg) * SS + keyg] == 0)
                    s[j][r] = -1.0e9f;
            }
        }
        // online softmax (rows m and m+8 live in disjoint 16-lane halves)
#pragma unroll
        for (int r = 0; r < 8; ++r) {
            float tm = fmaxf(fmaxf(s[0][r], s[1][r]), fmaxf(s[2][r], s[3][r]));
            tm = fmaxf(tm, __shfl_xor(tm, 1, 32));
            tm = fmaxf(tm, __shfl_xor(tm, 2, 32));
            tm = fmaxf(tm, __shfl_xor(tm, 4, 32));
            tm = fmaxf(tm, __shfl_xor(tm, 8, 32));
            float mn = fmaxf(M[r], tm);
            float sc = __expf(M[r] - mn);
            float rs = 0.0f;
#pragma unroll
            for (int j = 0; j < 4; ++j) {
                float p = __expf(s[j][r] - mn);
                s[j][r] = p;
                rs += p;
            }
            rs += __shfl_xor(rs, 1, 32);
            rs += __shfl_xor(rs, 2, 32);
            rs += __shfl_xor(rs, 4, 32);
            rs += __shfl_xor(rs, 8, 32);
            L[r] = L[r] * sc + rs;
            M[r] = mn;
#pragma unroll
            for (int j = 0; j < 4; ++j) o[j][r] *= sc;
        }
        // C-layout -> A-fragment layout via per-wave LDS strip
#pragma unroll
        for (int j = 0; j < 4; ++j)
#pragma unroll
            for (int r = 0; r < 8; ++r)
                Pw[(r + hi8) * 72 + j * 16 + l] = (_Float16)s[j][r];
        asm volatile("s_wait_dscnt 0" ::: "memory");   // wave-local LDS RAW
        // O += P * V
#pragma unroll
        for (int kk = 0; kk < 64; kk += 32) {
            v16h pa = load_frag_a(Pw, 72, kk, lane);
#pragma unroll
            for (int j = 0; j < 4; ++j) {
                v16h bv = load_frag_b(&Vt[(j * 16) * 72], 72, kk, lane);
                o[j] = wmma_f16(pa, bv, o[j]);
            }
        }
    }

    // write normalized output (f32 workspace, pre-Wo)
#pragma unroll
    for (int j = 0; j < 4; ++j)
#pragma unroll
        for (int r = 0; r < 8; ++r) {
            const int qg  = q0 + wid * 16 + r + hi8;
            const int col = h * DKK + j * 16 + l;
            O[(size_t)qg * DD + col] = o[j][r] / L[r];
        }
}

// ---------------------------------------------------------------------------
extern "C" void kernel_launch(void* const* d_in, const int* in_sizes, int n_in,
                              void* d_out, int out_size, void* d_ws, size_t ws_size,
                              hipStream_t stream) {
    (void)in_sizes; (void)n_in; (void)out_size; (void)ws_size;
    const float* query = (const float*)d_in[0];
    const float* key_  = (const float*)d_in[1];
    const float* value = (const float*)d_in[2];
    const int*   mask  = (const int*)d_in[3];
    const float* mz    = (const float*)d_in[4];
    const float* Wq = (const float*)d_in[5];
    const float* bq = (const float*)d_in[6];
    const float* Wk = (const float*)d_in[7];
    const float* bk = (const float*)d_in[8];
    const float* Wv = (const float*)d_in[9];
    const float* bv = (const float*)d_in[10];
    const float* Wo = (const float*)d_in[11];
    const float* bo = (const float*)d_in[12];
    float* out = (float*)d_out;

    // workspace layout: q16 | k16 | v16 (8 MiB f16 each) | attn f32 (16 MiB)
    char* ws = (char*)d_ws;
    const size_t MB = 1024 * 1024;
    _Float16* q16  = (_Float16*)(ws);
    _Float16* k16  = (_Float16*)(ws + 8  * MB);
    _Float16* v16  = (_Float16*)(ws + 16 * MB);
    float*    attn = (float*)   (ws + 24 * MB);

    dim3 blk(256);
    dim3 gproj(DD / 64, (BB * SS) / 128);          // 16 x 32
    proj_gemm<<<gproj, blk, 0, stream>>>(query, Wq, bq, mz, q16, nullptr, 1);
    proj_gemm<<<gproj, blk, 0, stream>>>(key_,  Wk, bk, mz, k16, nullptr, 1);
    proj_gemm<<<gproj, blk, 0, stream>>>(value, Wv, bv, mz, v16, nullptr, 0);
    attention<<<dim3(BB * HH * (SS / 128)), blk, 0, stream>>>(q16, k16, v16, mask, attn);
    proj_gemm<<<gproj, blk, 0, stream>>>(attn, Wo, bo, mz, nullptr, out, 2);
}
